// BiLSTM2_23965917511820
// MI455X (gfx1250) — matrix operand order repaired and checked by reading, not compiled, not run on hardware
//
#include <hip/hip_runtime.h>

typedef __attribute__((ext_vector_type(16))) _Float16 v16h;
typedef __attribute__((ext_vector_type(8)))  float    v8f;

#define BATCH  1024
#define TSTEPS 169
#define HID    128
#define GATES  512

__device__ __forceinline__ float tanh_fast(float x) {
#if __has_builtin(__builtin_amdgcn_tanhf)
    return __builtin_amdgcn_tanhf(x);   // V_TANH_F32 (TRANS pipe, co-executes with WMMA)
#else
    return tanhf(x);
#endif
}
__device__ __forceinline__ float sigmoid_fast(float x) {
    return 0.5f + 0.5f * tanh_fast(0.5f * x);
}

// ---------------------------------------------------------------------------
// Weight prep: packed [Wx;Wh]^T in f16, K zero-padded, and gate columns
// PERMUTED so that wave w, tile c, lane-col un  ->  gate c of unit u=16w+un.
//   n' = w*64 + c*16 + un   maps to original column  n = c*128 + (w*16 + un)
// This lets the LSTM cell run entirely in accumulator registers (no LDS gate
// round-trip).
// ---------------------------------------------------------------------------
__global__ void wprep_kernel(const float* __restrict__ Wx, const float* __restrict__ Wh,
                             _Float16* __restrict__ Wt, int kin, int kinpad, int ktot)
{
    int idx = blockIdx.x * blockDim.x + threadIdx.x;
    int total = GATES * ktot;
    if (idx >= total) return;
    int np = idx / ktot;               // permuted column
    int k  = idx - np * ktot;
    int w  = np >> 6;
    int c  = (np >> 4) & 3;
    int un = np & 15;
    int n  = c * 128 + w * 16 + un;    // original gate column
    float v;
    if (k < kinpad) v = (k < kin) ? Wx[(size_t)k * GATES + n] : 0.0f;
    else            v = Wh[(size_t)(k - kinpad) * GATES + n];
    Wt[(size_t)np * ktot + k] = (_Float16)v;
}

// ---------------------------------------------------------------------------
// One LSTM pass (one direction of one layer). Persistent WG per 16-row batch
// tile; serial over T; gates = [x_t | h] x Wt^T via v_wmma_f32_16x16x32_f16.
// Gate-permuted weights => cell phase fully in registers.
// ---------------------------------------------------------------------------
template<int KIN, int KINPAD, int KTOT, bool IN_F16>
__global__ __launch_bounds__(256) void lstm_pass_kernel(
    const void* __restrict__ in_v, const _Float16* __restrict__ Wt,
    const float* __restrict__ bias, const int* __restrict__ seqlen,
    _Float16* __restrict__ out, int out_c0, int reverse)
{
    __shared__ _Float16 Alds[16 * KTOT];   // [row][k]: x part [0,KINPAD), h part [KINPAD,KTOT)
    __shared__ int      Ls[16];            // per-row sequence lengths

    const int tid   = threadIdx.x;
    const int bbase = blockIdx.x * 16;
    const int wave  = tid >> 5;            // 0..7
    const int lane  = tid & 31;
    const int half  = lane >> 4;           // lane 0-15 vs 16-31
    const int mn    = lane & 15;
    const int u     = wave * 16 + mn;      // hidden unit owned by this lane (0..127)

    if (tid < 16) Ls[tid] = seqlen[bbase + tid];
    for (int i = tid; i < 16 * HID; i += 256) {
        int row = i >> 7, uu = i & (HID - 1);
        Alds[row * KTOT + KINPAD + uu] = (_Float16)0.0f;   // h0 = 0
    }
    // per-lane gate biases (original layout: i|j|f|o blocks of 128)
    const float bg0 = bias[u];
    const float bg1 = bias[u + 128];
    const float bg2 = bias[u + 256];
    const float bg3 = bias[u + 384];

    float cst[8];
    #pragma unroll
    for (int j = 0; j < 8; ++j) cst[j] = 0.0f;            // c0 = 0
    __syncthreads();

    int Lr[8];
    #pragma unroll
    for (int j = 0; j < 8; ++j) Lr[j] = Ls[j + 8 * half];

    const float*    inF = (const float*)in_v;
    const _Float16* inH = (const _Float16*)in_v;

    for (int t = 0; t < TSTEPS; ++t) {
        // -------- fill x part of A (per-row time index handles reverse_seq) ----
        for (int i = tid; i < 16 * KINPAD; i += 256) {
            int row = i / KINPAD;
            int kk  = i - row * KINPAD;
            int L   = Ls[row];
            int tin = reverse ? ((t < L) ? (L - 1 - t) : t) : t;
            int b   = bbase + row;
            _Float16 v = (_Float16)0.0f;
            if (kk < KIN) {
                size_t src = ((size_t)b * TSTEPS + tin) * KIN + kk;
                v = IN_F16 ? inH[src] : (_Float16)inF[src];
            }
            Alds[row * KTOT + kk] = v;
        }
        // prefetch next step's input rows while this step computes
        if (tid < 16 && t + 1 < TSTEPS) {
            int L   = Ls[tid];
            int tn  = t + 1;
            int tin = reverse ? ((tn < L) ? (L - 1 - tn) : tn) : tn;
            size_t src = ((size_t)(bbase + tid) * TSTEPS + tin) * KIN;
            __builtin_prefetch(IN_F16 ? (const void*)&inH[src] : (const void*)&inF[src], 0, 0);
        }
        __syncthreads();

        // -------- WMMA gate GEMM: 16 x 512, K = KTOT ---------------------------
        v8f acc[4];
        #pragma unroll
        for (int c = 0; c < 4; ++c)
            #pragma unroll
            for (int j = 0; j < 8; ++j) acc[c][j] = 0.0f;

        #pragma unroll
        for (int kc = 0; kc < KTOT; kc += 32) {
            // A fragment: 16-bit 16x32 layout (ISA 7.12.2)
            union { v16h v; unsigned uu[8]; } Af;
            #pragma unroll
            for (int j = 0; j < 8; ++j) {
                int k = (j < 4) ? (half * 8 + 2 * j) : (16 + half * 8 + 2 * (j - 4));
                Af.uu[j] = *(const unsigned*)&Alds[mn * KTOT + kc + k];
            }
            #pragma unroll
            for (int c = 0; c < 4; ++c) {
                int np = wave * 64 + c * 16 + mn;    // permuted weight row
                union { v16h v; unsigned uu[8]; } Bf;
                const _Float16* wp = Wt + (size_t)np * KTOT + kc + half * 8;
                #pragma unroll
                for (int j = 0; j < 8; ++j)
                    Bf.uu[j] = *(const unsigned*)&wp[(j < 4) ? (2 * j) : (2 * j + 8)];
                acc[c] = __builtin_amdgcn_wmma_f32_16x16x32_f16(
                    false, Af.v, false, Bf.v, (short)0, acc[c], false, false);
            }
        }
        __syncthreads();   // all waves done reading Alds before h is rewritten

        // -------- LSTM cell, fully in registers --------------------------------
        // acc[c][j] = gate c (i,j,f,o), batch row (j + 8*half), hidden unit u
        #pragma unroll
        for (int j = 0; j < 8; ++j) {
            float gi = acc[0][j] + bg0;
            float gj = acc[1][j] + bg1;
            float gf = acc[2][j] + bg2;
            float go = acc[3][j] + bg3;
            float cn = cst[j] * sigmoid_fast(gf + 1.0f) + sigmoid_fast(gi) * tanh_fast(gj);
            float hn = tanh_fast(cn) * sigmoid_fast(go);
            int   row = j + 8 * half;
            int   L   = Lr[j];
            bool  act = (t < L);
            if (act) cst[j] = cn;                          // carry state past seq_len
            int b   = bbase + row;
            int pos = act ? (reverse ? (L - 1 - t) : t) : t;
            out[((size_t)b * TSTEPS + pos) * 256 + out_c0 + u] = (_Float16)(act ? hn : 0.0f);
            if (act) Alds[row * KTOT + KINPAD + u] = (_Float16)hn;   // h for next step
        }
        __syncthreads();
    }
}

// ---------------------------------------------------------------------------
// Static branch: relu(relu(x_static @ w_s0 + b_s0) @ w_s1 + b_s1)
// ---------------------------------------------------------------------------
__global__ void static_branch_kernel(const float* __restrict__ xs,
                                     const float* __restrict__ w0, const float* __restrict__ b0,
                                     const float* __restrict__ w1, const float* __restrict__ b1,
                                     float* __restrict__ s_out)
{
    int row = blockIdx.x * blockDim.x + threadIdx.x;
    if (row >= BATCH) return;
    float h0[16];
    #pragma unroll
    for (int o = 0; o < 16; ++o) h0[o] = b0[o];
    #pragma unroll
    for (int k = 0; k < 19; ++k) {
        float xv = xs[row * 19 + k];
        #pragma unroll
        for (int o = 0; o < 16; ++o) h0[o] += xv * w0[k * 16 + o];
    }
    #pragma unroll
    for (int o = 0; o < 16; ++o) h0[o] = fmaxf(h0[o], 0.0f);
    float h1[16];
    #pragma unroll
    for (int o = 0; o < 16; ++o) h1[o] = b1[o];
    #pragma unroll
    for (int k = 0; k < 16; ++k) {
        float xv = h0[k];
        #pragma unroll
        for (int o = 0; o < 16; ++o) h1[o] += xv * w1[k * 16 + o];
    }
    #pragma unroll
    for (int o = 0; o < 16; ++o) s_out[row * 16 + o] = fmaxf(h1[o], 0.0f);
}

// ---------------------------------------------------------------------------
// Attention pooling: p = softmax_t(tanh(cur1 @ w_att + b)); att = sum_t p*cur1
// ---------------------------------------------------------------------------
__global__ __launch_bounds__(256) void attention_kernel(
    const _Float16* __restrict__ cur1, const float* __restrict__ w_att,
    const float* __restrict__ b_att, float* __restrict__ att_out)
{
    __shared__ float wlds[256];
    __shared__ float sc[256];
    __shared__ float red[256];
    int tid = threadIdx.x, b = blockIdx.x;
    wlds[tid] = w_att[tid];
    __syncthreads();
    float raw = -1e30f;
    if (tid < TSTEPS) {
        float s = b_att[0];
        const _Float16* p = cur1 + ((size_t)b * TSTEPS + tid) * 256;
        for (int d = 0; d < 256; ++d) s += (float)p[d] * wlds[d];
        raw = tanh_fast(s);
    }
    red[tid] = raw;
    for (int st = 128; st > 0; st >>= 1) { __syncthreads(); if (tid < st) red[tid] = fmaxf(red[tid], red[tid + st]); }
    __syncthreads();
    float mx = red[0];
    __syncthreads();
    float e = (tid < TSTEPS) ? __expf(raw - mx) : 0.0f;
    red[tid] = e;
    for (int st = 128; st > 0; st >>= 1) { __syncthreads(); if (tid < st) red[tid] += red[tid + st]; }
    __syncthreads();
    float denom = red[0];
    __syncthreads();
    sc[tid] = e / denom;
    __syncthreads();
    float att = 0.0f;
    for (int t = 0; t < TSTEPS; ++t)
        att += sc[t] * (float)cur1[((size_t)b * TSTEPS + t) * 256 + tid];
    att_out[(size_t)b * 256 + tid] = att;
}

// ---------------------------------------------------------------------------
// Classifier: relu(relu([s, att] @ w_c1 + b_c1) @ w_c2 + b_c2)
// ---------------------------------------------------------------------------
__global__ __launch_bounds__(256) void classifier_kernel(
    const float* __restrict__ s_in, const float* __restrict__ att,
    const float* __restrict__ w1, const float* __restrict__ b1,
    const float* __restrict__ w2, const float* __restrict__ b2,
    float* __restrict__ out)
{
    int row = blockIdx.x * blockDim.x + threadIdx.x;
    if (row >= BATCH) return;
    float h1[64];
    #pragma unroll
    for (int o = 0; o < 64; ++o) h1[o] = b1[o];
    for (int k = 0; k < 272; ++k) {
        float ck = (k < 16) ? s_in[row * 16 + k] : att[(size_t)row * 256 + (k - 16)];
        const float* wr = w1 + (size_t)k * 64;
        #pragma unroll
        for (int o = 0; o < 64; ++o) h1[o] += ck * wr[o];
    }
    float o2[32];
    #pragma unroll
    for (int q = 0; q < 32; ++q) o2[q] = b2[q];
    for (int o = 0; o < 64; ++o) {
        float v = fmaxf(h1[o], 0.0f);
        const float* wr = w2 + (size_t)o * 32;
        #pragma unroll
        for (int q = 0; q < 32; ++q) o2[q] += v * wr[q];
    }
    #pragma unroll
    for (int q = 0; q < 32; ++q) out[(size_t)row * 32 + q] = fmaxf(o2[q], 0.0f);
}

// ---------------------------------------------------------------------------
extern "C" void kernel_launch(void* const* d_in, const int* in_sizes, int n_in,
                              void* d_out, int out_size, void* d_ws, size_t ws_size,
                              hipStream_t stream)
{
    const float* x_static = (const float*)d_in[0];
    const float* x_dyn    = (const float*)d_in[1];
    const int*   seqlen   = (const int*)d_in[2];
    const float* w_s0 = (const float*)d_in[3];
    const float* b_s0 = (const float*)d_in[4];
    const float* w_s1 = (const float*)d_in[5];
    const float* b_s1 = (const float*)d_in[6];
    const float* Wx_f0 = (const float*)d_in[7];
    const float* Wh_f0 = (const float*)d_in[8];
    const float* bb_f0 = (const float*)d_in[9];
    const float* Wx_b0 = (const float*)d_in[10];
    const float* Wh_b0 = (const float*)d_in[11];
    const float* bb_b0 = (const float*)d_in[12];
    const float* Wx_f1 = (const float*)d_in[13];
    const float* Wh_f1 = (const float*)d_in[14];
    const float* bb_f1 = (const float*)d_in[15];
    const float* Wx_b1 = (const float*)d_in[16];
    const float* Wh_b1 = (const float*)d_in[17];
    const float* bb_b1 = (const float*)d_in[18];
    const float* w_att = (const float*)d_in[19];
    const float* b_att = (const float*)d_in[20];
    const float* w_c1 = (const float*)d_in[21];
    const float* b_c1 = (const float*)d_in[22];
    const float* w_c2 = (const float*)d_in[23];
    const float* b_c2 = (const float*)d_in[24];
    float* out = (float*)d_out;
    (void)in_sizes; (void)n_in; (void)out_size; (void)ws_size;

    char* ws = (char*)d_ws;
    size_t off = 0;
    auto alloc = [&](size_t bytes) -> char* {
        char* p = ws + off;
        off = (off + bytes + 255) & ~(size_t)255;
        return p;
    };
    _Float16* cur0  = (_Float16*)alloc((size_t)BATCH * TSTEPS * 256 * 2);
    _Float16* cur1  = (_Float16*)alloc((size_t)BATCH * TSTEPS * 256 * 2);
    _Float16* wt_f0 = (_Float16*)alloc((size_t)GATES * 192 * 2);
    _Float16* wt_b0 = (_Float16*)alloc((size_t)GATES * 192 * 2);
    _Float16* wt_f1 = (_Float16*)alloc((size_t)GATES * 384 * 2);
    _Float16* wt_b1 = (_Float16*)alloc((size_t)GATES * 384 * 2);
    float* s_ws   = (float*)alloc((size_t)BATCH * 16 * 4);
    float* att_ws = (float*)alloc((size_t)BATCH * 256 * 4);

    int tot0 = GATES * 192, tot1 = GATES * 384;
    wprep_kernel<<<(tot0 + 255) / 256, 256, 0, stream>>>(Wx_f0, Wh_f0, wt_f0, 36, 64, 192);
    wprep_kernel<<<(tot0 + 255) / 256, 256, 0, stream>>>(Wx_b0, Wh_b0, wt_b0, 36, 64, 192);
    wprep_kernel<<<(tot1 + 255) / 256, 256, 0, stream>>>(Wx_f1, Wh_f1, wt_f1, 256, 256, 384);
    wprep_kernel<<<(tot1 + 255) / 256, 256, 0, stream>>>(Wx_b1, Wh_b1, wt_b1, 256, 256, 384);

    static_branch_kernel<<<BATCH / 256, 256, 0, stream>>>(x_static, w_s0, b_s0, w_s1, b_s1, s_ws);

    lstm_pass_kernel<36, 64, 192, false><<<BATCH / 16, 256, 0, stream>>>(x_dyn, wt_f0, bb_f0, seqlen, cur0, 0,   0);
    lstm_pass_kernel<36, 64, 192, false><<<BATCH / 16, 256, 0, stream>>>(x_dyn, wt_b0, bb_b0, seqlen, cur0, 128, 1);
    lstm_pass_kernel<256, 256, 384, true><<<BATCH / 16, 256, 0, stream>>>(cur0, wt_f1, bb_f1, seqlen, cur1, 0,   0);
    lstm_pass_kernel<256, 256, 384, true><<<BATCH / 16, 256, 0, stream>>>(cur0, wt_b1, bb_b1, seqlen, cur1, 128, 1);

    attention_kernel<<<BATCH, 256, 0, stream>>>(cur1, w_att, b_att, att_ws);
    classifier_kernel<<<BATCH / 256, 256, 0, stream>>>(s_ws, att_ws, w_c1, b_c1, w_c2, b_c2, out);
}
